// MTREncoder_79293686218836
// MI455X (gfx1250) — compile-verified, matrix operand order checked by reading, not run on hardware
//
#include <hip/hip_runtime.h>
#include <hip/hip_bf16.h>
#include <math.h>

// ---------------- problem constants ----------------
#define B_    32
#define N_    64
#define T_    11
#define R_    2048
#define L_    16
#define TL_   11
#define G_    100
#define F_    128
#define DK_   64
#define HID_  256
#define S_TOT 2229                    // 1 + 64 + 2048 + 16 + 100
#define NROWS (B_ * S_TOT)            // 71328
#define KNN_K 8

typedef __attribute__((ext_vector_type(16))) __bf16 v16bf;
typedef __attribute__((ext_vector_type(8)))  float  v8f;

// ---------------- CDNA5 async global->LDS path (guarded; falls back to float4 copy) ----------------
#if __has_builtin(__builtin_amdgcn_global_load_async_to_lds_b128) && \
    __has_builtin(__builtin_amdgcn_s_wait_asynccnt)
#define HAVE_ASYNC_LDS 1
#endif

#ifdef HAVE_ASYNC_LDS
typedef int v4i_raw __attribute__((vector_size(16)));
typedef __attribute__((address_space(1))) v4i_raw* async_gptr;
typedef __attribute__((address_space(3))) v4i_raw* async_lptr;
#endif

__device__ __forceinline__ void lds_cp16(void* lds_dst, const void* gsrc) {
#ifdef HAVE_ASYNC_LDS
  __builtin_amdgcn_global_load_async_to_lds_b128(
      (async_gptr)gsrc, (async_lptr)lds_dst, 0, 0);
#else
  *reinterpret_cast<float4*>(lds_dst) = *reinterpret_cast<const float4*>(gsrc);
#endif
}
__device__ __forceinline__ void async_join() {
#ifdef HAVE_ASYNC_LDS
  __builtin_amdgcn_s_wait_asynccnt(0);
#endif
}

// ---------------- WMMA fragment helpers ----------------
// A (16x32 bf16): lanes 0-15 -> M=lane, K=kc*32+0..15 ; lanes 16-31 -> M=lane-16, K=kc*32+16..31
__device__ __forceinline__ v16bf wmma_load_a(const __bf16* base, int stride, int kc) {
  const int lane = threadIdx.x & 31;
  const int m    = lane & 15;
  const int kb   = kc * 32 + ((lane >> 4) << 4);
  return *reinterpret_cast<const v16bf*>(base + m * stride + kb);
}
// B: pre-swizzled so each lane reads 16 contiguous bf16 (32B) per (ntile,kchunk)
__device__ __forceinline__ v16bf wmma_load_b(const __bf16* W, int nt, int kc, int nK) {
  const int lane = threadIdx.x & 31;
  return *reinterpret_cast<const v16bf*>(W + ((((size_t)nt * nK + kc) * 32 + lane) << 4));
}
__device__ __forceinline__ v8f wmma_bf16(v16bf a, v16bf b, v8f c) {
  return __builtin_amdgcn_wmma_f32_16x16x32_bf16(false, a, false, b, (short)0, c, false, false);
}

__device__ __forceinline__ float gelu_tanh(float v) {
  const float c = 0.7978845608028654f;   // sqrt(2/pi)
  return 0.5f * v * (1.f + tanhf(c * (v + 0.044715f * v * v * v)));
}

// ---------------- weight swizzle: f32 [Kd,Nd] row-major -> B-fragment order bf16 ----------------
__global__ __launch_bounds__(256) void swizzleW(const float* __restrict__ src,
                                                __bf16* __restrict__ dst, int Kd, int Nd) {
  int tid = blockIdx.x * 256 + threadIdx.x;
  if (tid >= Kd * Nd) return;
  int j    = tid & 15;
  int lane = (tid >> 4) & 31;
  int rest = tid >> 9;
  int nK   = Kd >> 5;
  int kc   = rest % nK;
  int nt   = rest / nK;
  int k    = kc * 32 + ((lane >> 4) << 4) + j;
  int n    = nt * 16 + (lane & 15);
  dst[tid] = (__bf16)src[(size_t)k * Nd + n];
}

// ---------------- fused 3-layer MLP (128 -> 256 relu -> 256 relu -> 64) ----------------
// one wave owns a 32-row tile (two 16-row sub-tiles): each B fragment feeds 2 WMMAs,
// halving B-operand L2 traffic. No barriers: per-wave LDS, same-wave DS ops are in-order.
__global__ __launch_bounds__(32)
void mlp3_kernel(const float* __restrict__ X,
                 const __bf16* __restrict__ W0, const float* __restrict__ b0,
                 const __bf16* __restrict__ W1, const float* __restrict__ b1,
                 const __bf16* __restrict__ W2, const float* __restrict__ b2,
                 float* __restrict__ Out, int ntiles,
                 int rpb, int outStride, int outBase) {
  __shared__ __align__(32) __bf16 xT[32 * 128];
  __shared__ __align__(32) __bf16 h0[32 * 256];
  __shared__ __align__(32) __bf16 h1[32 * 256];
  const int lane = threadIdx.x;
  const int tile = blockIdx.x;
  if (tile >= ntiles) return;
  const int row0 = tile * 32;
  for (int i = lane; i < 32 * 128; i += 32)
    xT[i] = (__bf16)X[(size_t)row0 * 128 + i];      // rows are contiguous
  const int n     = lane & 15;
  const int mbase = (lane >> 4) << 3;
  // layer 0: 128 -> 256, relu
  for (int nt = 0; nt < 16; ++nt) {
    v8f c0, c1; float bi = b0[nt * 16 + n];
    #pragma unroll
    for (int v = 0; v < 8; ++v) { c0[v] = bi; c1[v] = bi; }
    #pragma unroll
    for (int kc = 0; kc < 4; ++kc) {
      v16bf b = wmma_load_b(W0, nt, kc, 4);
      c0 = wmma_bf16(wmma_load_a(xT, 128, kc), b, c0);
      c1 = wmma_bf16(wmma_load_a(xT + 16 * 128, 128, kc), b, c1);
    }
    #pragma unroll
    for (int v = 0; v < 8; ++v) {
      h0[(mbase + v) * 256 + nt * 16 + n]        = (__bf16)fmaxf(c0[v], 0.f);
      h0[(16 + mbase + v) * 256 + nt * 16 + n]   = (__bf16)fmaxf(c1[v], 0.f);
    }
  }
  // layer 1: 256 -> 256, relu
  for (int nt = 0; nt < 16; ++nt) {
    v8f c0, c1; float bi = b1[nt * 16 + n];
    #pragma unroll
    for (int v = 0; v < 8; ++v) { c0[v] = bi; c1[v] = bi; }
    #pragma unroll
    for (int kc = 0; kc < 8; ++kc) {
      v16bf b = wmma_load_b(W1, nt, kc, 8);
      c0 = wmma_bf16(wmma_load_a(h0, 256, kc), b, c0);
      c1 = wmma_bf16(wmma_load_a(h0 + 16 * 256, 256, kc), b, c1);
    }
    #pragma unroll
    for (int v = 0; v < 8; ++v) {
      h1[(mbase + v) * 256 + nt * 16 + n]        = (__bf16)fmaxf(c0[v], 0.f);
      h1[(16 + mbase + v) * 256 + nt * 16 + n]   = (__bf16)fmaxf(c1[v], 0.f);
    }
  }
  // layer 2: 256 -> 64, with row remap into output
  for (int nt = 0; nt < 4; ++nt) {
    v8f c0, c1; float bi = b2[nt * 16 + n];
    #pragma unroll
    for (int v = 0; v < 8; ++v) { c0[v] = bi; c1[v] = bi; }
    #pragma unroll
    for (int kc = 0; kc < 8; ++kc) {
      v16bf b = wmma_load_b(W2, nt, kc, 8);
      c0 = wmma_bf16(wmma_load_a(h1, 256, kc), b, c0);
      c1 = wmma_bf16(wmma_load_a(h1 + 16 * 256, 256, kc), b, c1);
    }
    #pragma unroll
    for (int v = 0; v < 8; ++v) {
      int r0 = row0 + mbase + v, r1 = r0 + 16;
      int bb0 = r0 / rpb, ii0 = r0 % rpb;
      int bb1 = r1 / rpb, ii1 = r1 % rpb;
      Out[((size_t)bb0 * outStride + outBase + ii0) * 64 + nt * 16 + n] = c0[v];
      Out[((size_t)bb1 * outStride + outBase + ii1) * 64 + nt * 16 + n] = c1[v];
    }
  }
}

// ---------------- generic single linear layer (Kd<=64), used for q/k/v projections ----------------
__global__ __launch_bounds__(64)
void linear_kernel(const float* __restrict__ X, const __bf16* __restrict__ W,
                   const float* __restrict__ bias, float* __restrict__ Out,
                   int ntiles, int Kd, int Nd) {
  __shared__ __align__(32) __bf16 xT[2][16 * 64];
  const int w    = threadIdx.x >> 5;
  const int lane = threadIdx.x & 31;
  const int tile = blockIdx.x * 2 + w;
  if (tile >= ntiles) return;
  const int row0 = tile * 16;
  for (int i = lane; i < 16 * Kd; i += 32) {
    int r = i / Kd, c = i % Kd;
    xT[w][r * 64 + c] = (__bf16)X[(size_t)(row0 + r) * Kd + c];
  }
  const int n = lane & 15, mbase = (lane >> 4) << 3;
  const int nK = Kd >> 5, nT = Nd >> 4;
  for (int nt = 0; nt < nT; ++nt) {
    v8f c; float bi = bias[nt * 16 + n];
    #pragma unroll
    for (int v = 0; v < 8; ++v) c[v] = bi;
    for (int kc = 0; kc < nK; ++kc)
      c = wmma_bf16(wmma_load_a(&xT[w][0], 64, kc), wmma_load_b(W, nt, kc, nK), c);
    #pragma unroll
    for (int v = 0; v < 8; ++v)
      Out[(size_t)(row0 + mbase + v) * Nd + nt * 16 + n] = c[v];
  }
}

// ---------------- feed-forward block: latent += alpha * (gelu(latent@W0+b0)@W1 + b1) ----------------
__global__ __launch_bounds__(64)
void ff_kernel(float* __restrict__ latent,
               const __bf16* __restrict__ W0, const float* __restrict__ b0,
               const __bf16* __restrict__ W1, const float* __restrict__ b1,
               const float* __restrict__ alphaPtr, int ntiles) {
  __shared__ __align__(32) __bf16 xT[2][16 * 64];
  __shared__ __align__(32) __bf16 h[2][16 * 256];
  const int w    = threadIdx.x >> 5;
  const int lane = threadIdx.x & 31;
  const int tile = blockIdx.x * 2 + w;
  if (tile >= ntiles) return;
  const float alpha = alphaPtr[0];
  const int row0 = tile * 16;
  for (int i = lane; i < 16 * 64; i += 32) {
    int r = i >> 6, c = i & 63;
    xT[w][i] = (__bf16)latent[(size_t)(row0 + r) * 64 + c];
  }
  const int n = lane & 15, mbase = (lane >> 4) << 3;
  for (int nt = 0; nt < 16; ++nt) {          // 64 -> 256, gelu
    v8f c; float bi = b0[nt * 16 + n];
    #pragma unroll
    for (int v = 0; v < 8; ++v) c[v] = bi;
    #pragma unroll
    for (int kc = 0; kc < 2; ++kc)
      c = wmma_bf16(wmma_load_a(&xT[w][0], 64, kc), wmma_load_b(W0, nt, kc, 2), c);
    #pragma unroll
    for (int v = 0; v < 8; ++v)
      h[w][(mbase + v) * 256 + nt * 16 + n] = (__bf16)gelu_tanh(c[v]);
  }
  for (int nt = 0; nt < 4; ++nt) {           // 256 -> 64, residual
    v8f c; float bi = b1[nt * 16 + n];
    #pragma unroll
    for (int v = 0; v < 8; ++v) c[v] = bi;
    #pragma unroll
    for (int kc = 0; kc < 8; ++kc)
      c = wmma_bf16(wmma_load_a(&h[w][0], 256, kc), wmma_load_b(W1, nt, kc, 8), c);
    #pragma unroll
    for (int v = 0; v < 8; ++v)
      latent[(size_t)(row0 + mbase + v) * 64 + nt * 16 + n] += alpha * c[v];
  }
}

// ---------------- masked maxpool + PE add + concat into latent; also builds x copy and masks ----------------
__global__ __launch_bounds__(64)
void pool_kernel(const float* __restrict__ stage,
                 const unsigned char* __restrict__ sdcM, const unsigned char* __restrict__ othM,
                 const unsigned char* __restrict__ rgM,  const unsigned char* __restrict__ tlM,
                 const float* __restrict__ peS, const float* __restrict__ peO,
                 const float* __restrict__ peR, const float* __restrict__ peT,
                 const float* __restrict__ peG,
                 float* __restrict__ latent, float* __restrict__ x,
                 int* __restrict__ lmask, int* __restrict__ xmask) {
  const int row = blockIdx.x;               // b*S + s
  const int b = row / S_TOT, s = row % S_TOT, d = threadIdx.x;
  const float* stOth = stage + (size_t)(B_ * T_) * 64;                 // after sdc rows
  const float* stTl  = stage + (size_t)(B_ * T_ + B_ * N_ * T_) * 64;  // after other rows
  float val; int m;
  if (s == 0) {
    float mx = -INFINITY; int any = 0;
    for (int t = 0; t < T_; ++t)
      if (sdcM[b * T_ + t]) { mx = fmaxf(mx, stage[((size_t)b * T_ + t) * 64 + d]); any = 1; }
    val = (any ? mx : 0.f) + peS[d]; m = any;
  } else if (s < 1 + N_) {
    int nn = s - 1;
    float mx = -INFINITY; int any = 0;
    for (int t = 0; t < T_; ++t)
      if (othM[(b * N_ + nn) * T_ + t]) { mx = fmaxf(mx, stOth[((size_t)(b * N_ + nn) * T_ + t) * 64 + d]); any = 1; }
    val = (any ? mx : 0.f) + peO[nn * 64 + d]; m = any;
    x[((size_t)b * N_ + nn) * 64 + d] = val;
    if (d == 0) xmask[b * N_ + nn] = m;
  } else if (s < 1 + N_ + R_) {
    int r = s - (1 + N_);
    val = latent[(size_t)row * 64 + d] + peR[r * 64 + d];
    m = rgM[b * R_ + r];
  } else if (s < 1 + N_ + R_ + L_) {
    int l = s - (1 + N_ + R_);
    float mx = -INFINITY; int any = 0;
    for (int t = 0; t < TL_; ++t)
      if (tlM[(b * L_ + l) * TL_ + t]) { mx = fmaxf(mx, stTl[((size_t)(b * L_ + l) * TL_ + t) * 64 + d]); any = 1; }
    val = (any ? mx : 0.f) + peT[l * 64 + d]; m = any;
  } else {
    int g = s - (1 + N_ + R_ + L_);
    val = latent[(size_t)row * 64 + d] + peG[g * 64 + d];
    m = 1;
  }
  latent[(size_t)row * 64 + d] = val;
  if (d == 0) lmask[row] = m;
}

// ---------------- KNN: per (b,s) top-8 nearest x rows (|x|^2 - 2 lat.x ranking, stable ties) ----------------
__global__ __launch_bounds__(256)
void knn_kernel(const float* __restrict__ latent, const float* __restrict__ x,
                const int* __restrict__ xmask, int* __restrict__ idxOut) {
  __shared__ __align__(16) float sx[N_ * 64];
  __shared__ float sn[N_];
  __shared__ int   sm[N_];
  const int b = blockIdx.y;
  // async bulk fill of x[b] (16KB) into LDS
  for (int i = threadIdx.x; i < (N_ * 64) / 4; i += 256)
    lds_cp16(&sx[i * 4], x + (size_t)b * N_ * 64 + i * 4);
  async_join();
  __syncthreads();
  for (int i = threadIdx.x; i < N_; i += 256) {
    float s = 0.f;
    for (int d = 0; d < 64; ++d) { float v = sx[i * 64 + d]; s += v * v; }
    sn[i] = s; sm[i] = xmask[b * N_ + i];
  }
  __syncthreads();
  const int s = blockIdx.x * 256 + threadIdx.x;
  if (s >= S_TOT) return;
  const float* lp = latent + ((size_t)b * S_TOT + s) * 64;
  float lr[64];
  #pragma unroll
  for (int d = 0; d < 64; ++d) lr[d] = lp[d];
  float bd[KNN_K]; int bi[KNN_K];
  #pragma unroll
  for (int j = 0; j < KNN_K; ++j) { bd[j] = 1e38f; bi[j] = 0; }
  for (int j = 0; j < N_; ++j) {
    float dot = 0.f;
    for (int t = 0; t < 64; ++t) dot += lr[t] * sx[j * 64 + t];
    float d2 = sm[j] ? (sn[j] - 2.f * dot) : 1e37f;
    if (d2 < bd[KNN_K - 1]) {
      int p = KNN_K - 1;
      while (p > 0 && d2 < bd[p - 1]) { bd[p] = bd[p - 1]; bi[p] = bi[p - 1]; --p; }
      bd[p] = d2; bi[p] = j;
    }
  }
  #pragma unroll
  for (int j = 0; j < KNN_K; ++j) idxOut[((size_t)b * S_TOT + s) * KNN_K + j] = bi[j];
}

// ---------------- local attention: gather 8 neighbors, softmax(2 heads x 16), Wo proj, residual ----------------
__global__ __launch_bounds__(256)
void attn_kernel(const float* __restrict__ q, const float* __restrict__ kk,
                 const float* __restrict__ vv, const int* __restrict__ idx,
                 const int* __restrict__ xmask, const int* __restrict__ lmask,
                 const float* __restrict__ Wo, const float* __restrict__ bo,
                 const float* __restrict__ alphaPtr, float* __restrict__ latent,
                 int useQmask) {
  __shared__ __align__(16) float sWo[32 * 64];
  __shared__ __align__(16) float sbo[64];
  for (int i = threadIdx.x; i < 512; i += 256)
    lds_cp16(&sWo[i * 4], Wo + i * 4);
  if (threadIdx.x < 16)
    lds_cp16(&sbo[threadIdx.x * 4], bo + threadIdx.x * 4);
  async_join();
  __syncthreads();
  const int tid = blockIdx.x * 256 + threadIdx.x;
  if (tid >= NROWS) return;
  const int b = tid / S_TOT;
  const float alpha = alphaPtr[0];
  float qv[32];
  #pragma unroll
  for (int i = 0; i < 32; ++i) qv[i] = q[(size_t)tid * 32 + i];
  int nb[KNN_K];
  #pragma unroll
  for (int j = 0; j < KNN_K; ++j) nb[j] = idx[(size_t)tid * KNN_K + j];
  float o[32];
  #pragma unroll
  for (int i = 0; i < 32; ++i) o[i] = 0.f;
  for (int h = 0; h < 2; ++h) {
    float sc[KNN_K];
    for (int j = 0; j < KNN_K; ++j) {
      const float* kp = kk + ((size_t)b * N_ + nb[j]) * 32 + h * 16;
      float d = 0.f;
      #pragma unroll
      for (int t = 0; t < 16; ++t) d += qv[h * 16 + t] * kp[t];
      sc[j] = xmask[b * N_ + nb[j]] ? d * 0.25f : -1e30f;   // 1/sqrt(16)
    }
    float mx = sc[0];
    for (int j = 1; j < KNN_K; ++j) mx = fmaxf(mx, sc[j]);
    float e[KNN_K], den = 0.f;
    for (int j = 0; j < KNN_K; ++j) { e[j] = __expf(sc[j] - mx); den += e[j]; }
    float inv = 1.f / den;
    for (int j = 0; j < KNN_K; ++j) {
      float wgt = e[j] * inv;
      const float* vp = vv + ((size_t)b * N_ + nb[j]) * 32 + h * 16;
      #pragma unroll
      for (int t = 0; t < 16; ++t) o[h * 16 + t] += wgt * vp[t];
    }
  }
  float qm = 1.f;
  if (useQmask) qm = lmask[tid] ? 1.f : 0.f;
  const float scale = alpha * qm;
  for (int d = 0; d < 64; ++d) {
    float acc = sbo[d];
    #pragma unroll
    for (int i = 0; i < 32; ++i) acc += o[i] * sWo[i * 64 + d];
    latent[(size_t)tid * 64 + d] += scale * acc;
  }
}

// ---------------- final mean over S ----------------
__global__ __launch_bounds__(64)
void mean_kernel(const float* __restrict__ latent, float* __restrict__ out) {
  const int b = blockIdx.x, d = threadIdx.x;
  float s = 0.f;
  for (int i = 0; i < S_TOT; ++i) s += latent[((size_t)b * S_TOT + i) * 64 + d];
  out[b * 64 + d] = s / (float)S_TOT;
}

// ================================================================ host
extern "C" void kernel_launch(void* const* d_in, const int* in_sizes, int n_in,
                              void* d_out, int out_size, void* d_ws, size_t ws_size,
                              hipStream_t stream) {
  (void)in_sizes; (void)n_in; (void)out_size; (void)ws_size;
  const int P = 9;  // params start (flattened dict order)
  auto FP = [&](int i) { return (const float*)d_in[i]; };
  const unsigned char* sdcM = (const unsigned char*)d_in[5];
  const unsigned char* othM = (const unsigned char*)d_in[6];
  const unsigned char* rgM  = (const unsigned char*)d_in[7];
  const unsigned char* tlM  = (const unsigned char*)d_in[8];

  // workspace layout (256B aligned)
  char* ws = (char*)d_ws;
  size_t off = 0;
  auto alloc = [&](size_t bytes) -> char* {
    char* p = ws + off;
    off += (bytes + 255) & ~(size_t)255;
    return p;
  };
  float* latent = (float*)alloc((size_t)NROWS * 64 * 4);
  float* xbuf   = (float*)alloc((size_t)B_ * N_ * 64 * 4);
  float* qbuf   = (float*)alloc((size_t)NROWS * 32 * 4);
  float* kkbuf  = (float*)alloc((size_t)B_ * N_ * 32 * 4);
  float* vvbuf  = (float*)alloc((size_t)B_ * N_ * 32 * 4);
  int*   idxbuf = (int*)alloc((size_t)NROWS * KNN_K * 4);
  int*   lmask  = (int*)alloc((size_t)NROWS * 4);
  int*   xmask  = (int*)alloc((size_t)B_ * N_ * 4);
  const size_t stageRows = (size_t)B_ * T_ + (size_t)B_ * N_ * T_ + (size_t)B_ * L_ * TL_;
  float* stage  = (float*)alloc(stageRows * 64 * 4);
  __bf16* arena = (__bf16*)alloc((size_t)(5 * 114688 + 12 * 2048 + 3 * 32768) * 2);

  // swizzle+convert all weights that feed WMMA
  size_t aoff = 0;
  auto swz = [&](int pidx, int Kd, int Nd) -> const __bf16* {
    __bf16* dst = arena + aoff;
    aoff += (size_t)Kd * Nd;
    int total = Kd * Nd;
    swizzleW<<<(total + 255) / 256, 256, 0, stream>>>(FP(pidx), dst, Kd, Nd);
    return dst;
  };
  // mlp sets: sdc, other, rg, tl, gps -> param base P + set*6
  const __bf16 *mW0[5], *mW1[5], *mW2[5];
  for (int i = 0; i < 5; ++i) {
    mW0[i] = swz(P + i * 6 + 0, F_, HID_);
    mW1[i] = swz(P + i * 6 + 2, HID_, HID_);
    mW2[i] = swz(P + i * 6 + 4, HID_, DK_);
  }
  auto attnBase = [&](int i) { return P + 35 + (i == 0 ? 0 : 9 + (i - 1) * 13); };
  const __bf16 *Wq[4], *Wk[4], *Wv[4], *fW0[4], *fW1[4];
  for (int i = 0; i < 4; ++i) {
    int ab = attnBase(i);
    Wq[i] = swz(ab + 0, DK_, 32);
    Wk[i] = swz(ab + 2, DK_, 32);
    Wv[i] = swz(ab + 4, DK_, 32);
    if (i > 0) {
      fW0[i] = swz(ab + 9, DK_, 256);
      fW1[i] = swz(ab + 11, 256, DK_);
    }
  }

  // ---- 5 fused MLPs (one wave per 32-row tile) ----
  const float* stSdc = stage;
  float* stSdcW = stage;
  float* stOthW = stage + (size_t)B_ * T_ * 64;
  float* stTlW  = stage + ((size_t)B_ * T_ + (size_t)B_ * N_ * T_) * 64;
  auto mlp = [&](const float* X, int set, float* out, int nrows, int rpb, int ostride, int obase) {
    int ntiles = nrows / 32;
    mlp3_kernel<<<ntiles, 32, 0, stream>>>(
        X, mW0[set], FP(P + set * 6 + 1), mW1[set], FP(P + set * 6 + 3),
        mW2[set], FP(P + set * 6 + 5), out, ntiles, rpb, ostride, obase);
  };
  mlp(FP(0), 0, stSdcW, B_ * T_,        B_ * T_,        0,     0);      // sdc -> stage
  mlp(FP(1), 1, stOthW, B_ * N_ * T_,   B_ * N_ * T_,   0,     0);      // other -> stage
  mlp(FP(2), 2, latent, B_ * R_,        R_,             S_TOT, 1 + N_); // rg -> latent slots
  mlp(FP(3), 3, stTlW,  B_ * L_ * TL_,  B_ * L_ * TL_,  0,     0);      // tl -> stage
  mlp(FP(4), 4, latent, B_ * G_,        G_,             S_TOT, 1 + N_ + R_ + L_); // gps -> latent

  // ---- pool + PE + concat ----
  pool_kernel<<<NROWS, 64, 0, stream>>>(stSdc, sdcM, othM, rgM, tlM,
                                        FP(P + 30), FP(P + 31), FP(P + 32), FP(P + 33), FP(P + 34),
                                        latent, xbuf, lmask, xmask);

  // ---- 4 attention layers (+ ff for layers 1..3) ----
  const int qTiles = NROWS / 16;          // 4458
  const int kvTiles = (B_ * N_) / 16;     // 128
  for (int i = 0; i < 4; ++i) {
    int ab = attnBase(i);
    knn_kernel<<<dim3((S_TOT + 255) / 256, B_), 256, 0, stream>>>(latent, xbuf, xmask, idxbuf);
    linear_kernel<<<(qTiles + 1) / 2, 64, 0, stream>>>(latent, Wq[i], FP(ab + 1), qbuf, qTiles, 64, 32);
    linear_kernel<<<(kvTiles + 1) / 2, 64, 0, stream>>>(xbuf, Wk[i], FP(ab + 3), kkbuf, kvTiles, 64, 32);
    linear_kernel<<<(kvTiles + 1) / 2, 64, 0, stream>>>(xbuf, Wv[i], FP(ab + 5), vvbuf, kvTiles, 64, 32);
    attn_kernel<<<(NROWS + 255) / 256, 256, 0, stream>>>(qbuf, kkbuf, vvbuf, idxbuf, xmask, lmask,
                                                         FP(ab + 6), FP(ab + 7), FP(ab + 8), latent,
                                                         i == 0 ? 1 : 0);
    if (i > 0)
      ff_kernel<<<(qTiles + 1) / 2, 64, 0, stream>>>(latent, fW0[i], FP(ab + 10), fW1[i], FP(ab + 12),
                                                     FP(ab + 8), qTiles);
  }

  // ---- final mean over S ----
  mean_kernel<<<B_, 64, 0, stream>>>(latent, (float*)d_out);
}
